// Seq2Seq_GNN_Transformer_35682588295658
// MI455X (gfx1250) — matrix operand order repaired and checked by reading, compiled, not run on hardware
//
#include <hip/hip_runtime.h>
#include <hip/hip_bf16.h>

// Seq2Seq GNN Transformer for MI455X (gfx1250, wave32, WMMA).
// All heavy GEMMs use v_wmma_f32_16x16x32_f16 (f16 in, f32 accumulate).
// All B operands live in LDS in K-contiguous (transposed) layout so every
// fragment load is two ds_load_b128 (no scalar ds_load_u16 + v_mov_b16 packing).
// Workspace layout (d_ws, needs ROWS*(64+96)*2 = ~120 MB):
//   A1: ROWS x 64 f16  ([xg(17)|agg(17)|pad]) -> reused as enc_x after fc_in
//   A2: ROWS x 96 f16  (enc_fc_in input, 89 pad 96) -> reused as enc_out
// Input pointer order assumed: setup_inputs() dict insertion order, params
// flattened in insertion order (43 tensors total).

#define BB 16
#define NN 1024
#define TT 24
#define SS (BB*NN)          // 16384 sequences
#define ROWS (SS*TT)        // 393216 matrix rows

typedef _Float16 v8h  __attribute__((ext_vector_type(8)));
typedef _Float16 v16h __attribute__((ext_vector_type(16)));
typedef float    v8f  __attribute__((ext_vector_type(8)));

static __device__ __forceinline__ v8f wmma_f16(v16h a, v16h b, v8f c) {
  return __builtin_amdgcn_wmma_f32_16x16x32_f16(false, a, false, b, (short)0, c, false, false);
}

// ---- WMMA fragment loaders (CDNA5 wave32 layouts, ISA 7.12.2) ----
// A 16x32 f16: lane row = m0+(lane&15); group g=lane>>4; f[0..7]=K k0+g*8+0..7,
// f[8..15]=K k0+16+g*8+0..7  (two contiguous 16B chunks -> vector loads)
static __device__ __forceinline__ v16h frag_a32(const _Float16* A, int ld, int m0, int k0) {
  int lane = threadIdx.x & 31;
  const _Float16* p = A + (m0 + (lane & 15)) * ld + k0 + ((lane >> 4) << 3);
  v8h lo = *(const v8h*)p;
  v8h hi = *(const v8h*)(p + 16);
  return __builtin_shufflevector(lo, hi, 0,1,2,3,4,5,6,7,8,9,10,11,12,13,14,15);
}
// A fragment with only 16 real K values (K 16..31 zero)
static __device__ __forceinline__ v16h frag_a16(const _Float16* A, int ld, int m0, int k0) {
  int lane = threadIdx.x & 31;
  const _Float16* p = A + (m0 + (lane & 15)) * ld + k0 + ((lane >> 4) << 3);
  v8h lo = *(const v8h*)p;
  v8h hi = {};
  return __builtin_shufflevector(lo, hi, 0,1,2,3,4,5,6,7,8,9,10,11,12,13,14,15);
}
// B 32x16 f16 from K-CONTIGUOUS storage BT[n][k] (BT[n*ldk + k] = B[k][n]):
// lane col = n0+(lane&15); group g: f[0..7]=K k0+g*8+0..7, f[8..15]=K k0+16+g*8+0..7 -> 2 vector loads
static __device__ __forceinline__ v16h frag_bT32(const _Float16* BT, int ldk, int k0, int n0) {
  int lane = threadIdx.x & 31;
  const _Float16* p = BT + (n0 + (lane & 15)) * ldk + k0 + ((lane >> 4) << 3);
  v8h lo = *(const v8h*)p;
  v8h hi = *(const v8h*)(p + 16);
  return __builtin_shufflevector(lo, hi, 0,1,2,3,4,5,6,7,8,9,10,11,12,13,14,15);
}
// Same but only 16 real K rows (K 16..31 zero). Also serves as B = Kmat^T
// when Kmat is row-major [N][Kdim] (that storage is already K-contiguous).
static __device__ __forceinline__ v16h frag_bT16(const _Float16* BT, int ldk, int k0, int n0) {
  int lane = threadIdx.x & 31;
  const _Float16* p = BT + (n0 + (lane & 15)) * ldk + k0 + ((lane >> 4) << 3);
  v8h lo = *(const v8h*)p;
  v8h hi = {};
  return __builtin_shufflevector(lo, hi, 0,1,2,3,4,5,6,7,8,9,10,11,12,13,14,15);
}
// C/D: element r -> row m0 + (lane>>4)*8 + r, col n0 + (lane&15)
static __device__ __forceinline__ void store_h_bias(_Float16* D, int ld, int m0, int n0,
                                                    v8f c, const float* bias) {
  int lane = threadIdx.x & 31;
  int col = n0 + (lane & 15);
  int row = m0 + ((lane >> 4) << 3);
  float bv = bias[col];
  #pragma unroll
  for (int r = 0; r < 8; ++r) D[(row + r)*ld + col] = (_Float16)(c[r] + bv);
}
static __device__ __forceinline__ void store_h(_Float16* D, int ld, int m0, int n0, v8f c) {
  int lane = threadIdx.x & 31;
  int col = n0 + (lane & 15);
  int row = m0 + ((lane >> 4) << 3);
  #pragma unroll
  for (int r = 0; r < 8; ++r) D[(row + r)*ld + col] = (_Float16)c[r];
}
// Store D tile TRANSPOSED (DT[col*ldk + row]); the 8 rows per lane are
// consecutive -> single packed ds_store_b128 per call.
static __device__ __forceinline__ void store_hT_bias(_Float16* DT, int ldk, int m0, int n0,
                                                     v8f c, const float* bias) {
  int lane = threadIdx.x & 31;
  int col = n0 + (lane & 15);
  int row = m0 + ((lane >> 4) << 3);
  float bv = bias[col];
  v8h h;
  #pragma unroll
  for (int r = 0; r < 8; ++r) h[r] = (_Float16)(c[r] + bv);
  *(v8h*)(DT + col*ldk + row) = h;
}
static __device__ __forceinline__ void store_sc(float* Sc, int m0, int n0, v8f c, float scale) {
  int lane = threadIdx.x & 31;
  int col = n0 + (lane & 15);
  int row = m0 + ((lane >> 4) << 3);
  #pragma unroll
  for (int r = 0; r < 8; ++r) Sc[(row + r)*32 + col] = c[r] * scale;
}
// row-wise softmax over 24 cols; Sc 32x32 f32 (overwritten), Pt 32x32 f16
static __device__ __forceinline__ void softmax24(float* Sc, _Float16* Pt) {
  int r = threadIdx.x & 31;
  if (r < 24) {
    float mx = -1e30f;
    for (int c = 0; c < 24; ++c) mx = fmaxf(mx, Sc[r*32 + c]);
    float sum = 0.f;
    for (int c = 0; c < 24; ++c) { float e = __expf(Sc[r*32 + c] - mx); Sc[r*32 + c] = e; sum += e; }
    float inv = 1.0f / sum;
    for (int c = 0; c < 24; ++c) Pt[r*32 + c] = (_Float16)(Sc[r*32 + c] * inv);
    for (int c = 24; c < 32; ++c) Pt[r*32 + c] = (_Float16)0;
  } else {
    for (int c = 0; c < 32; ++c) Pt[r*32 + c] = (_Float16)0;
  }
}

// ---------------- kernel 1: GCN gather + pack f16 A matrices ----------------
__global__ void __launch_bounds__(256) k_prep(
    const float* __restrict__ X, const float* __restrict__ Y,
    const float* __restrict__ EW, const int* __restrict__ EIDX,
    const float* __restrict__ pos_enc,
    _Float16* __restrict__ A1, _Float16* __restrict__ A2) {
  int s = blockIdx.x;            // node id = b*N + n
  int b = s >> 10, n = s & 1023;
  int tid = threadIdx.x;
  __shared__ int   ssrc[8];
  __shared__ float swt[8];
  if (tid < 8) { ssrc[tid] = EIDX[s*8 + tid]; swt[tid] = EW[s*8 + tid]; }
  __syncthreads();
  // xg (col 0 = y, 1..16 = X) and agg over 8 weighted neighbors
  for (int e = tid; e < TT*17; e += 256) {
    int t = e / 17, k = e % 17;
    int bt = (b*48 + t) * 1024;
    float xv = (k == 0) ? Y[bt + n] : X[(bt + n)*16 + (k-1)];
    float acc = 0.f;
    #pragma unroll
    for (int j = 0; j < 8; ++j) {
      int sp = ssrc[j];
      int nb = sp & 1023;
      int btn = ((sp >> 10)*48 + t) * 1024;
      float nv = (k == 0) ? Y[btn + nb] : X[(btn + nb)*16 + (k-1)];
      acc += swt[j] * nv;
    }
    int row = s*TT + t;
    A1[row*64 + k]      = (_Float16)xv;
    A1[row*64 + 17 + k] = (_Float16)acc;
  }
  // zero-pad A1 cols 34..63
  for (int e = tid; e < TT*30; e += 256) {
    int t = e / 30, c = 34 + e % 30;
    A1[(s*TT + t)*64 + c] = (_Float16)0;
  }
  // A2 cols: [0..7]=pos_enc, [8..71]=h (kernel 2), [72..87]=Xh, [88]=yh, [89..95]=0
  for (int e = tid; e < TT*32; e += 256) {
    int t = e >> 5, c = e & 31;
    int row = s*TT + t;
    int bt = (b*48 + t) * 1024;
    _Float16 v;
    int col;
    if (c < 8)       { v = (_Float16)pos_enc[t*8 + c];           col = c; }
    else if (c < 24) { v = (_Float16)X[(bt + n)*16 + (c - 8)];   col = c + 64; }
    else if (c == 24){ v = (_Float16)Y[bt + n];                  col = 88; }
    else             { v = (_Float16)0;                          col = c + 64; }
    A2[row*96 + col] = v;
  }
}

// ------------- kernel 2: h = sigmoid([xg|agg] @ [Wroot;Wrel] + b) -----------
__global__ void __launch_bounds__(256) k_gemm_gcn(
    const float* __restrict__ w_root, const float* __restrict__ w_rel,
    const float* __restrict__ gcn_b,
    const _Float16* __restrict__ A1, _Float16* __restrict__ A2) {
  __shared__ _Float16 sWT[64*64];       // [n][k] transposed
  __shared__ float sB[64];
  int tid = threadIdx.x;
  for (int i = tid; i < 64*64; i += 256) {
    int nn2 = i >> 6, k = i & 63;
    float v = (k < 17) ? w_root[k*64 + nn2] : (k < 34) ? w_rel[(k-17)*64 + nn2] : 0.0f;
    sWT[i] = (_Float16)v;
  }
  if (tid < 64) sB[tid] = gcn_b[tid];
  __syncthreads();
  int m0 = (blockIdx.x * 8 + (tid >> 5)) * 16;
  v16h a0 = frag_a32(A1, 64, m0, 0);
  v16h a1 = frag_a32(A1, 64, m0, 32);
  int lane = tid & 31;
  int colL = lane & 15, rb = (lane >> 4) << 3;
  for (int nt = 0; nt < 4; ++nt) {
    v8f c = {};
    c = wmma_f16(a0, frag_bT32(sWT, 64, 0,  nt*16), c);
    c = wmma_f16(a1, frag_bT32(sWT, 64, 32, nt*16), c);
    int col = nt*16 + colL;
    float bv = sB[col];
    #pragma unroll
    for (int r = 0; r < 8; ++r) {
      float v = c[r] + bv;
      v = 1.0f / (1.0f + __expf(-v));                 // sigmoid
      A2[(m0 + rb + r)*96 + 8 + col] = (_Float16)v;
    }
  }
}

// --------------- kernel 3: enc_x = A2(89 pad 96) @ Wfc + b ------------------
__global__ void __launch_bounds__(256) k_gemm_fcin(
    const float* __restrict__ Wfc, const float* __restrict__ bfc,
    const _Float16* __restrict__ A2, _Float16* __restrict__ EX) {
  __shared__ _Float16 sWT[64*96];       // [n][k] transposed, k pad 96
  __shared__ float sB[64];
  int tid = threadIdx.x;
  for (int i = tid; i < 64*96; i += 256) {
    int nn2 = i / 96, k = i % 96;
    sWT[i] = (k < 89) ? (_Float16)Wfc[k*64 + nn2] : (_Float16)0;
  }
  if (tid < 64) sB[tid] = bfc[tid];
  __syncthreads();
  int m0 = (blockIdx.x * 8 + (tid >> 5)) * 16;
  v16h a0 = frag_a32(A2, 96, m0, 0);
  v16h a1 = frag_a32(A2, 96, m0, 32);
  v16h a2 = frag_a32(A2, 96, m0, 64);
  for (int nt = 0; nt < 4; ++nt) {
    v8f c = {};
    c = wmma_f16(a0, frag_bT32(sWT, 96, 0,  nt*16), c);
    c = wmma_f16(a1, frag_bT32(sWT, 96, 32, nt*16), c);
    c = wmma_f16(a2, frag_bT32(sWT, 96, 64, nt*16), c);
    store_h_bias(EX, 64, m0, nt*16, c, sB);
  }
}

// ------ kernel 4: encoder MHA + mlp; 1 wave = 1 sequence (T=24 pad 32) ------
__global__ void __launch_bounds__(128) k_encoder(
    const _Float16* __restrict__ EX,
    const float* __restrict__ wq, const float* __restrict__ bq,
    const float* __restrict__ wk, const float* __restrict__ bk,
    const float* __restrict__ wv, const float* __restrict__ bv,
    const float* __restrict__ wo, const float* __restrict__ bo,
    const float* __restrict__ wm, const float* __restrict__ bm,
    _Float16* __restrict__ EO) {
  extern __shared__ char smem[];
  _Float16* sW = (_Float16*)smem;                 // 5 x 64x64, each [n][k]
  float* sB = (float*)(smem + 5*4096*2);          // 5 x 64
  char* pseq = smem + 5*4096*2 + 5*64*4;
  int tid = threadIdx.x, wave = tid >> 5, lane = tid & 31;
  for (int i = tid; i < 5*4096; i += 128) {
    int m = i >> 12, j = i & 4095;
    int nn2 = j >> 6, k = j & 63;
    const float* sp = (m==0)?wq:(m==1)?wk:(m==2)?wv:(m==3)?wo:wm;
    sW[i] = (_Float16)sp[k*64 + nn2];             // transposed
  }
  if (tid < 64) {
    sB[tid] = bq[tid]; sB[64+tid] = bk[tid]; sB[128+tid] = bv[tid];
    sB[192+tid] = bo[tid]; sB[256+tid] = bm[tid];
  }
  __syncthreads();
  int s = blockIdx.x * 4 + wave;
  _Float16* Xt = (_Float16*)(pseq + wave * 26624);  // 32x64 row-major, later Z
  _Float16* Qt = Xt + 2048;                         // 32x64 row-major
  _Float16* Kt = Qt + 2048;                         // 32x64 row-major (=K^T B)
  _Float16* VT = Kt + 2048;                         // 64x32 TRANSPOSED V
  _Float16* AO = VT + 2048;                         // 32x64 row-major
  float*    Sc = (float*)(AO + 2048);               // 32x32 f32
  _Float16* Pt = (_Float16*)(Sc + 1024);            // 32x32 f16
  for (int i = lane; i < 2048; i += 32) {
    int t = i >> 6, c = i & 63;
    Xt[i] = (t < 24) ? EX[(s*TT + t)*64 + c] : (_Float16)0;
  }
  __syncthreads();
  // Q/K projections (row-major out), V projection (transposed out)
  for (int p = 0; p < 3; ++p) {
    const _Float16* W = sW + p*4096;
    const float* bb = sB + p*64;
    for (int mt = 0; mt < 2; ++mt) {
      v16h a0 = frag_a32(Xt, 64, mt*16, 0);
      v16h a1 = frag_a32(Xt, 64, mt*16, 32);
      for (int nt = 0; nt < 4; ++nt) {
        v8f c = {};
        c = wmma_f16(a0, frag_bT32(W, 64, 0,  nt*16), c);
        c = wmma_f16(a1, frag_bT32(W, 64, 32, nt*16), c);
        if (p == 2)      store_hT_bias(VT, 32, mt*16, nt*16, c, bb);
        else if (p == 1) store_h_bias(Kt, 64, mt*16, nt*16, c, bb);
        else             store_h_bias(Qt, 64, mt*16, nt*16, c, bb);
      }
    }
  }
  __syncthreads();
  // attention, 4 heads, dh=16, scale 1/4
  for (int h = 0; h < 4; ++h) {
    int hc = h << 4;
    for (int mt = 0; mt < 2; ++mt) {
      v16h a = frag_a16(Qt, 64, mt*16, hc);
      for (int nt = 0; nt < 2; ++nt) {
        v8f c = {};
        c = wmma_f16(a, frag_bT16(Kt, 64, hc, nt*16), c);   // B = K^T
        store_sc(Sc, mt*16, nt*16, c, 0.25f);
      }
    }
    __syncthreads();
    softmax24(Sc, Pt);
    __syncthreads();
    for (int mt = 0; mt < 2; ++mt) {
      v16h a = frag_a32(Pt, 32, mt*16, 0);     // P cols 24..31 are zero
      v8f c = {};
      c = wmma_f16(a, frag_bT32(VT, 32, 0, hc), c);
      store_h(AO, 64, mt*16, hc, c);
    }
    __syncthreads();
  }
  // out-projection Wo -> Xt (reuse)
  for (int mt = 0; mt < 2; ++mt) {
    v16h a0 = frag_a32(AO, 64, mt*16, 0);
    v16h a1 = frag_a32(AO, 64, mt*16, 32);
    for (int nt = 0; nt < 4; ++nt) {
      v8f c = {};
      c = wmma_f16(a0, frag_bT32(sW + 3*4096, 64, 0,  nt*16), c);
      c = wmma_f16(a1, frag_bT32(sW + 3*4096, 64, 32, nt*16), c);
      store_h_bias(Xt, 64, mt*16, nt*16, c, sB + 192);
    }
  }
  __syncthreads();
  // mlp -> enc_out (global, rows < 24 only)
  for (int mt = 0; mt < 2; ++mt) {
    v16h a0 = frag_a32(Xt, 64, mt*16, 0);
    v16h a1 = frag_a32(Xt, 64, mt*16, 32);
    for (int nt = 0; nt < 4; ++nt) {
      v8f c = {};
      c = wmma_f16(a0, frag_bT32(sW + 4*4096, 64, 0,  nt*16), c);
      c = wmma_f16(a1, frag_bT32(sW + 4*4096, 64, 32, nt*16), c);
      int col = nt*16 + (lane & 15);
      int r0 = mt*16 + ((lane >> 4) << 3);
      float bv = sB[256 + col];
      #pragma unroll
      for (int r = 0; r < 8; ++r) {
        int row = r0 + r;
        if (row < 24) EO[(s*TT + row)*64 + col] = (_Float16)(c[r] + bv);
      }
    }
  }
}

// --------- kernel 5: decoder (fc_in, causal SA, fc_ed, cross-MHA, out) ------
__global__ void __launch_bounds__(64) k_decoder(
    const float* __restrict__ X, const float* __restrict__ pos_dec,
    const _Float16* __restrict__ EO,
    const float* __restrict__ fin_w, const float* __restrict__ fin_b,
    const float* __restrict__ sa_wq, const float* __restrict__ sa_bq,
    const float* __restrict__ sa_wk, const float* __restrict__ sa_bk,
    const float* __restrict__ sa_wv, const float* __restrict__ sa_bv,
    const float* __restrict__ sa_wo, const float* __restrict__ sa_bo,
    const float* __restrict__ fed_w, const float* __restrict__ fed_b,
    const float* __restrict__ ca_wq, const float* __restrict__ ca_bq,
    const float* __restrict__ ca_wk, const float* __restrict__ ca_bk,
    const float* __restrict__ ca_wv, const float* __restrict__ ca_bv,
    const float* __restrict__ ca_wo, const float* __restrict__ ca_bo,
    const float* __restrict__ out_w, const float* __restrict__ out_b,
    float* __restrict__ OUT) {
  extern __shared__ char smem[];
  _Float16* sDW  = (_Float16*)smem;        // 4 x 16x16 (sa q,k,v,o) [n][k]
  _Float16* sFin = sDW + 1024;             // 16n x 32k (11 pad 32) [n][k]
  _Float16* sFed = sFin + 512;             // 64n x 16k [n][k]
  _Float16* sCW  = sFed + 1024;            // 4 x 64x64 (ca q,k,v,o) [n][k]
  float* fB = (float*)(sCW + 16384);       // 512 floats of biases + out_w/b
  char* pseq = smem + 37888 + 2048;
  int tid = threadIdx.x, wave = tid >> 5, lane = tid & 31;
  for (int i = tid; i < 1024; i += 64) {
    int m = i >> 8, j = i & 255;
    int nn2 = j >> 4, k = j & 15;
    const float* sp = (m==0)?sa_wq:(m==1)?sa_wk:(m==2)?sa_wv:sa_wo;
    sDW[i] = (_Float16)sp[k*16 + nn2];     // transposed
  }
  for (int i = tid; i < 512; i += 64) {
    int nn2 = i >> 5, k = i & 31;
    sFin[i] = (k < 11) ? (_Float16)fin_w[k*16 + nn2] : (_Float16)0;  // transposed
  }
  for (int i = tid; i < 1024; i += 64) {
    int nn2 = i >> 4, k = i & 15;
    sFed[i] = (_Float16)fed_w[k*64 + nn2];                            // transposed
  }
  for (int i = tid; i < 16384; i += 64) {
    int m = i >> 12, j = i & 4095;
    int nn2 = j >> 6, k = j & 63;
    const float* sp = (m==0)?ca_wq:(m==1)?ca_wk:(m==2)?ca_wv:ca_wo;
    sCW[i] = (_Float16)sp[k*64 + nn2];                                // transposed
  }
  if (tid < 16) {
    fB[tid] = fin_b[tid];
    fB[16+tid] = sa_bq[tid]; fB[32+tid] = sa_bk[tid];
    fB[48+tid] = sa_bv[tid]; fB[64+tid] = sa_bo[tid];
  }
  if (tid < 64) {
    fB[80+tid]  = fed_b[tid];
    fB[144+tid] = ca_bq[tid]; fB[208+tid] = ca_bk[tid];
    fB[272+tid] = ca_bv[tid]; fB[336+tid] = ca_bo[tid];
    fB[400+tid] = out_w[tid];
  }
  if (tid == 0) fB[464] = out_b[0];
  __syncthreads();
  int s = blockIdx.x * 2 + wave;
  int b = s >> 10, n = s & 1023;
  char* my = pseq + wave * 33792;
  _Float16* Din = (_Float16*)my;   // 32x32 (dec_in K-padded), later dSA (32x16)
  _Float16* D0  = Din + 1024;      // 32x16
  _Float16* DQ  = D0 + 512;
  _Float16* DK  = DQ + 512;
  _Float16* DV  = DK + 512;
  _Float16* DAO = DV + 512;
  _Float16* D1  = DAO + 512;       // 32x64, later Z
  _Float16* Et  = D1 + 2048;       // enc tile, later CQ (row-major)
  _Float16* CK  = Et + 2048;       // 32x64 row-major (B = K^T source)
  _Float16* CVT = CK + 2048;       // 64x32 TRANSPOSED V
  _Float16* CAO = CVT + 2048;      // 32x64 row-major
  float*    Sc  = (float*)(CAO + 2048);
  _Float16* Pt  = (_Float16*)(Sc + 1024);
  // dec_in: [Xd(3) | pos_dec(8) | pad]
  for (int i = lane; i < 1024; i += 32) {
    int t = i >> 5, c = i & 31;
    _Float16 v = (_Float16)0;
    if (t < 24) {
      if (c < 3)       v = (_Float16)X[((b*48 + 24 + t)*1024 + n)*16 + 13 + c];
      else if (c < 11) v = (_Float16)pos_dec[t*8 + (c - 3)];
    }
    Din[i] = v;
  }
  for (int i = lane; i < 2048; i += 32) {
    int t = i >> 6, c = i & 63;
    Et[i] = (t < 24) ? EO[(s*TT + t)*64 + c] : (_Float16)0;
  }
  __syncthreads();
  // D0 = dec_in @ fc_in + b
  for (int mt = 0; mt < 2; ++mt) {
    v16h a = frag_a32(Din, 32, mt*16, 0);
    v8f c = {};
    c = wmma_f16(a, frag_bT32(sFin, 32, 0, 0), c);
    store_h_bias(D0, 16, mt*16, 0, c, fB);
  }
  __syncthreads();
  // self-attn projections (16x16)
  for (int p = 0; p < 3; ++p) {
    _Float16* Ot = (p==0)?DQ:(p==1)?DK:DV;
    for (int mt = 0; mt < 2; ++mt) {
      v16h a = frag_a16(D0, 16, mt*16, 0);
      v8f c = {};
      c = wmma_f16(a, frag_bT16(sDW + p*256, 16, 0, 0), c);
      store_h_bias(Ot, 16, mt*16, 0, c, fB + 16 + p*16);
    }
  }
  __syncthreads();
  // causal self-attention (heads=4, dh=4, scale 1/2) on VALU
  {
    int r = lane;
    if (r < 24) {
      for (int h = 0; h < 4; ++h) {
        float mx = -1e30f;
        for (int c2 = 0; c2 <= r; ++c2) {
          float sd = 0.f;
          #pragma unroll
          for (int j = 0; j < 4; ++j)
            sd += (float)DQ[r*16 + h*4 + j] * (float)DK[c2*16 + h*4 + j];
          sd *= 0.5f;
          Sc[r*32 + c2] = sd;
          mx = fmaxf(mx, sd);
        }
        float sum = 0.f;
        for (int c2 = 0; c2 <= r; ++c2) {
          float e = __expf(Sc[r*32 + c2] - mx);
          Sc[r*32 + c2] = e; sum += e;
        }
        float inv = 1.0f / sum;
        float o0 = 0.f, o1 = 0.f, o2 = 0.f, o3 = 0.f;
        for (int c2 = 0; c2 <= r; ++c2) {
          float pc = Sc[r*32 + c2] * inv;
          o0 += pc * (float)DV[c2*16 + h*4 + 0];
          o1 += pc * (float)DV[c2*16 + h*4 + 1];
          o2 += pc * (float)DV[c2*16 + h*4 + 2];
          o3 += pc * (float)DV[c2*16 + h*4 + 3];
        }
        DAO[r*16 + h*4 + 0] = (_Float16)o0; DAO[r*16 + h*4 + 1] = (_Float16)o1;
        DAO[r*16 + h*4 + 2] = (_Float16)o2; DAO[r*16 + h*4 + 3] = (_Float16)o3;
      }
    } else {
      for (int c2 = 0; c2 < 16; ++c2) DAO[r*16 + c2] = (_Float16)0;
    }
  }
  __syncthreads();
  // dSA = DAO @ sa_wo + bo  (into Din region, ld 16)
  for (int mt = 0; mt < 2; ++mt) {
    v16h a = frag_a16(DAO, 16, mt*16, 0);
    v8f c = {};
    c = wmma_f16(a, frag_bT16(sDW + 3*256, 16, 0, 0), c);
    store_h_bias(Din, 16, mt*16, 0, c, fB + 64);
  }
  __syncthreads();
  // D1 = dSA @ fc_ed + b
  for (int mt = 0; mt < 2; ++mt) {
    v16h a = frag_a16(Din, 16, mt*16, 0);
    for (int nt = 0; nt < 4; ++nt) {
      v8f c = {};
      c = wmma_f16(a, frag_bT16(sFed, 16, 0, nt*16), c);
      store_h_bias(D1, 64, mt*16, nt*16, c, fB + 80);
    }
  }
  __syncthreads();
  // CK (row-major) / CV (transposed) from enc tile
  for (int p = 0; p < 2; ++p) {
    int wi = p + 1;                       // wk=1, wv=2
    for (int mt = 0; mt < 2; ++mt) {
      v16h a0 = frag_a32(Et, 64, mt*16, 0);
      v16h a1 = frag_a32(Et, 64, mt*16, 32);
      for (int nt = 0; nt < 4; ++nt) {
        v8f c = {};
        c = wmma_f16(a0, frag_bT32(sCW + wi*4096, 64, 0,  nt*16), c);
        c = wmma_f16(a1, frag_bT32(sCW + wi*4096, 64, 32, nt*16), c);
        if (p == 0) store_h_bias(CK, 64, mt*16, nt*16, c, fB + 144 + wi*64);
        else        store_hT_bias(CVT, 32, mt*16, nt*16, c, fB + 144 + wi*64);
      }
    }
  }
  __syncthreads();
  // CQ = D1 @ ca_wq + bq  (into Et, enc tile no longer needed)
  for (int mt = 0; mt < 2; ++mt) {
    v16h a0 = frag_a32(D1, 64, mt*16, 0);
    v16h a1 = frag_a32(D1, 64, mt*16, 32);
    for (int nt = 0; nt < 4; ++nt) {
      v8f c = {};
      c = wmma_f16(a0, frag_bT32(sCW, 64, 0,  nt*16), c);
      c = wmma_f16(a1, frag_bT32(sCW, 64, 32, nt*16), c);
      store_h_bias(Et, 64, mt*16, nt*16, c, fB + 144);
    }
  }
  __syncthreads();
  // cross-attention heads (dh=16, scale 1/4)
  for (int h = 0; h < 4; ++h) {
    int hc = h << 4;
    for (int mt = 0; mt < 2; ++mt) {
      v16h a = frag_a16(Et, 64, mt*16, hc);
      for (int nt = 0; nt < 2; ++nt) {
        v8f c = {};
        c = wmma_f16(a, frag_bT16(CK, 64, hc, nt*16), c);   // B = K^T
        store_sc(Sc, mt*16, nt*16, c, 0.25f);
      }
    }
    __syncthreads();
    softmax24(Sc, Pt);
    __syncthreads();
    for (int mt = 0; mt < 2; ++mt) {
      v16h a = frag_a32(Pt, 32, mt*16, 0);
      v8f c = {};
      c = wmma_f16(a, frag_bT32(CVT, 32, 0, hc), c);
      store_h(CAO, 64, mt*16, hc, c);
    }
    __syncthreads();
  }
  // Z = CAO @ ca_wo + bo  (into D1)
  for (int mt = 0; mt < 2; ++mt) {
    v16h a0 = frag_a32(CAO, 64, mt*16, 0);
    v16h a1 = frag_a32(CAO, 64, mt*16, 32);
    for (int nt = 0; nt < 4; ++nt) {
      v8f c = {};
      c = wmma_f16(a0, frag_bT32(sCW + 3*4096, 64, 0,  nt*16), c);
      c = wmma_f16(a1, frag_bT32(sCW + 3*4096, 64, 32, nt*16), c);
      store_h_bias(D1, 64, mt*16, nt*16, c, fB + 336);
    }
  }
  __syncthreads();
  // preds = Z @ out_w + out_b  -> (B, FC, N, 1)
  if (lane < 24) {
    float acc = fB[464];
    for (int j = 0; j < 64; ++j) acc += (float)D1[lane*64 + j] * fB[400 + j];
    OUT[(b*TT + lane)*1024 + n] = acc;
  }
}

extern "C" void kernel_launch(void* const* d_in, const int* in_sizes, int n_in,
                              void* d_out, int out_size, void* d_ws, size_t ws_size,
                              hipStream_t stream) {
  (void)in_sizes; (void)n_in; (void)out_size; (void)ws_size;
  const float* X  = (const float*)d_in[0];
  const float* Y  = (const float*)d_in[1];
  const float* EW = (const float*)d_in[2];
  const float* gcn_w_root  = (const float*)d_in[3];
  const float* gcn_w_rel   = (const float*)d_in[4];
  const float* gcn_b       = (const float*)d_in[5];
  const float* enc_fc_in_w = (const float*)d_in[6];
  const float* enc_fc_in_b = (const float*)d_in[7];
  const float* enc_wq = (const float*)d_in[8];
  const float* enc_bq = (const float*)d_in[9];
  const float* enc_wk = (const float*)d_in[10];
  const float* enc_bk = (const float*)d_in[11];
  const float* enc_wv = (const float*)d_in[12];
  const float* enc_bv = (const float*)d_in[13];
  const float* enc_wo = (const float*)d_in[14];
  const float* enc_bo = (const float*)d_in[15];
  const float* enc_mlp_w = (const float*)d_in[16];
  const float* enc_mlp_b = (const float*)d_in[17];
  const float* pos_enc = (const float*)d_in[18];
  const float* pos_dec = (const float*)d_in[19];
  const float* dec_fc_in_w = (const float*)d_in[20];
  const float* dec_fc_in_b = (const float*)d_in[21];
  const float* dec_sa_wq = (const float*)d_in[22];
  const float* dec_sa_bq = (const float*)d_in[23];
  const float* dec_sa_wk = (const float*)d_in[24];
  const float* dec_sa_bk = (const float*)d_in[25];
  const float* dec_sa_wv = (const float*)d_in[26];
  const float* dec_sa_bv = (const float*)d_in[27];
  const float* dec_sa_wo = (const float*)d_in[28];
  const float* dec_sa_bo = (const float*)d_in[29];
  const float* dec_fc_ed_w = (const float*)d_in[30];
  const float* dec_fc_ed_b = (const float*)d_in[31];
  const float* dec_ca_wq = (const float*)d_in[32];
  const float* dec_ca_bq = (const float*)d_in[33];
  const float* dec_ca_wk = (const float*)d_in[34];
  const float* dec_ca_bk = (const float*)d_in[35];
  const float* dec_ca_wv = (const float*)d_in[36];
  const float* dec_ca_bv = (const float*)d_in[37];
  const float* dec_ca_wo = (const float*)d_in[38];
  const float* dec_ca_bo = (const float*)d_in[39];
  const float* dec_out_w = (const float*)d_in[40];
  const float* dec_out_b = (const float*)d_in[41];
  const int* EIDX = (const int*)d_in[42];

  char* ws = (char*)d_ws;
  _Float16* A1 = (_Float16*)ws;                        // ROWS x 64 -> later enc_x
  _Float16* A2 = (_Float16*)(ws + (size_t)ROWS*64*2);  // ROWS x 96 -> later enc_out

  k_prep<<<SS, 256, 0, stream>>>(X, Y, EW, EIDX, pos_enc, A1, A2);
  k_gemm_gcn<<<ROWS/128, 256, 0, stream>>>(gcn_w_root, gcn_w_rel, gcn_b, A1, A2);
  k_gemm_fcin<<<ROWS/128, 256, 0, stream>>>(enc_fc_in_w, enc_fc_in_b, A2, A1);
  size_t shmE = (size_t)(5*4096*2 + 5*64*4 + 4*26624);   // 148736 B
  k_encoder<<<SS/4, 128, shmE, stream>>>(A1, enc_wq, enc_bq, enc_wk, enc_bk,
                                         enc_wv, enc_bv, enc_wo, enc_bo,
                                         enc_mlp_w, enc_mlp_b, A2);
  size_t shmF = (size_t)(37888 + 2048 + 2*33792);        // 107520 B
  k_decoder<<<SS/2, 64, shmF, stream>>>(X, pos_dec, A2,
      dec_fc_in_w, dec_fc_in_b,
      dec_sa_wq, dec_sa_bq, dec_sa_wk, dec_sa_bk, dec_sa_wv, dec_sa_bv,
      dec_sa_wo, dec_sa_bo,
      dec_fc_ed_w, dec_fc_ed_b,
      dec_ca_wq, dec_ca_bq, dec_ca_wk, dec_ca_bk, dec_ca_wv, dec_ca_bv,
      dec_ca_wo, dec_ca_bo,
      dec_out_w, dec_out_b, (float*)d_out);
}
